// CausalSelfAttention_54949811585626
// MI455X (gfx1250) — matrix operand order repaired and checked by reading, not compiled, not run on hardware
//
#include <hip/hip_runtime.h>
#include <hip/hip_bf16.h>

#define B_  4
#define T_  2048
#define H_  1024
#define NH_ 16
#define HD_ 64
#define BT_ (B_ * T_)

typedef __bf16 bf16_t;
typedef __attribute__((ext_vector_type(16))) __bf16 v16bf;
typedef __attribute__((ext_vector_type(8)))  __bf16 v8bf;
typedef __attribute__((ext_vector_type(8)))  float  v8f;

union V16U { v16bf v; v8bf h[2]; };

__device__ __forceinline__ v8f wmma_bf16(v16bf a, v16bf b, v8f c) {
  // D = A(16x32 bf16) * B(32x16 bf16) + C(16x16 f32)
  return __builtin_amdgcn_wmma_f32_16x16x32_bf16(false, a, false, b, (short)0, c,
                                                 false, false);
}

// Async DMA: global -> LDS, 16 bytes per lane, tracked by ASYNCcnt.
// GVS addressing: mem = SGPR base + 32-bit per-lane byte offset.
// lds_addr = raw LDS byte offset (low 32 bits of the generic LDS address).
__device__ __forceinline__ void async_b128(unsigned lds_addr, unsigned goff,
                                           const void* sbase) {
  asm volatile("global_load_async_to_lds_b128 %0, %1, %2"
               :: "v"(lds_addr), "v"(goff), "s"(sbase)
               : "memory");
}
__device__ __forceinline__ void wait_async0() {
  asm volatile("s_wait_asynccnt 0x0" ::: "memory");
}
__device__ __forceinline__ unsigned lds_off(const void* p) {
  return (unsigned)(unsigned long long)p;   // aperture[63:32] | offset[31:0]
}

// A fragment (16x32 bf16) from a row-major tile (ld = elements per row).
// ISA 7.12.2: lane l holds row l%16; lanes 0-15: K in [0,8)+[16,24),
// lanes 16-31: K in [8,16)+[24,32); 2 bf16 per VGPR.
__device__ __forceinline__ v16bf load_a_frag(const bf16_t* base, int ld, int lane) {
  const int m  = lane & 15;
  const int c0 = (lane < 16) ? 0 : 8;
  const bf16_t* p = base + m * ld + c0;
  V16U u;
  u.h[0] = *(const v8bf*)(p);        // K = c0 .. c0+7
  u.h[1] = *(const v8bf*)(p + 16);   // K = c0+16 .. c0+23
  return u.v;
}

// B fragment (32x16 bf16) from an N-major tile: row n holds 32 contiguous K.
// ISA: lane l holds column n=l%16; lanes 0-15: K [0,8)+[16,24), lanes 16-31: K [8,16)+[24,32).
__device__ __forceinline__ v16bf load_b_frag(const bf16_t* base, int ld, int lane) {
  const int n  = lane & 15;
  const int kb = (lane < 16) ? 0 : 8;
  const bf16_t* p = base + n * ld + kb;
  V16U u;
  u.h[0] = *(const v8bf*)(p);        // K = kb .. kb+7
  u.h[1] = *(const v8bf*)(p + 16);   // K = kb+16 .. kb+23
  return u.v;
}

// ---------------------------------------------------------------- fp32 -> bf16
__global__ __launch_bounds__(256) void cvt_f32_to_bf16(
    const float* __restrict__ s, bf16_t* __restrict__ d, int n) {
  int i = (blockIdx.x * blockDim.x + threadIdx.x) * 4;
  if (i + 3 < n) {
    float4 f = *(const float4*)(s + i);
    d[i + 0] = (bf16_t)f.x;
    d[i + 1] = (bf16_t)f.y;
    d[i + 2] = (bf16_t)f.z;
    d[i + 3] = (bf16_t)f.w;
  }
}

// ---------------------------------------------------------------- QKV GEMM
// C[m,n] = sum_k X[m,k] * W[n,k] + bias[n]   (torch Linear: x @ W.T + b)
// Block: 256 threads = 8 waves; block tile 128(M) x 128(N); wave tile 32x64.
// Double-buffered async-to-LDS staging: DMA of tile k+1 overlaps WMMA of tile k.
__global__ __launch_bounds__(256) void qkv_gemm(
    const bf16_t* __restrict__ X,
    const bf16_t* __restrict__ W0, const bf16_t* __restrict__ W1,
    const bf16_t* __restrict__ W2,
    const float* __restrict__ b0, const float* __restrict__ b1,
    const float* __restrict__ b2,
    bf16_t* __restrict__ Q, bf16_t* __restrict__ K, bf16_t* __restrict__ V) {
  const int z = blockIdx.z;
  const bf16_t* W    = (z == 0) ? W0 : (z == 1) ? W1 : W2;
  const float*  bias = (z == 0) ? b0 : (z == 1) ? b1 : b2;
  bf16_t*       Out  = (z == 0) ? Q  : (z == 1) ? K  : V;

  // row stride 56 bf16 = 112 B: 16B-aligned, 28-bank stagger (16 distinct rows)
  __shared__ bf16_t Xs[2][128][56];
  __shared__ bf16_t Ws[2][128][56];

  const int tid  = threadIdx.x;
  const int lane = tid & 31;
  const int wave = tid >> 5;
  const int wm = wave & 3;            // 4 waves along M (4*32  = 128)
  const int wn = wave >> 2;           // 2 waves along N (2*64  = 128)
  const int mBase = blockIdx.x * 128;
  const int nBase = blockIdx.y * 128;

  v8f acc[2][4];
#pragma unroll
  for (int i = 0; i < 2; ++i)
#pragma unroll
    for (int j = 0; j < 4; ++j) acc[i][j] = (v8f){0, 0, 0, 0, 0, 0, 0, 0};

  const int lr = tid >> 1;            // 0..127: tile row this thread fills
  const int lc = (tid & 1) * 16;      // 0|16:   16 elems = 32 B per thread

  auto issue_tiles = [&](int kt, int bufi) {
    const unsigned gx = (unsigned)(((mBase + lr) * H_ + kt + lc) * 2);
    async_b128(lds_off(&Xs[bufi][lr][lc]),     gx,      X);
    async_b128(lds_off(&Xs[bufi][lr][lc + 8]), gx + 16, X);
    const unsigned gw = (unsigned)(((nBase + lr) * H_ + kt + lc) * 2);
    async_b128(lds_off(&Ws[bufi][lr][lc]),     gw,      W);
    async_b128(lds_off(&Ws[bufi][lr][lc + 8]), gw + 16, W);
  };

  issue_tiles(0, 0);
  for (int kt = 0; kt < H_; kt += 32) {
    const int cur = (kt >> 5) & 1;
    wait_async0();        // my DMAs for buf[cur] have landed
    __syncthreads();      // everyone's landed; everyone done reading buf[cur^1]
    if (kt + 32 < H_) issue_tiles(kt + 32, cur ^ 1);

#pragma unroll
    for (int ti = 0; ti < 2; ++ti) {
      v16bf a = load_a_frag(&Xs[cur][wm * 32 + ti * 16][0], 56, lane);
#pragma unroll
      for (int tj = 0; tj < 4; ++tj) {
        v16bf b = load_b_frag(&Ws[cur][wn * 64 + tj * 16][0], 56, lane);
        acc[ti][tj] = wmma_bf16(a, b, acc[ti][tj]);
      }
    }
  }

  // Epilogue: bias add, convert to bf16, store (C layout: lane<16 rows 0-7).
#pragma unroll
  for (int ti = 0; ti < 2; ++ti)
#pragma unroll
    for (int tj = 0; tj < 4; ++tj) {
      const int n = nBase + wn * 64 + tj * 16 + (lane & 15);
      const float bv = bias[n];
      const int m0 = mBase + wm * 32 + ti * 16 + ((lane < 16) ? 0 : 8);
#pragma unroll
      for (int r = 0; r < 8; ++r)
        Out[(size_t)(m0 + r) * H_ + n] = (bf16_t)(acc[ti][tj][r] + bv);
    }
}

// ---------------------------------------------------------------- Flash attention
// Grid: (T/128 query tiles, NH, B). Block: 256 threads = 8 waves.
// Each wave owns a 16-query strip: private online softmax, no cross-wave reduce.
__global__ __launch_bounds__(256) void attn_fwd(
    const bf16_t* __restrict__ Q, const bf16_t* __restrict__ K,
    const bf16_t* __restrict__ V, const float* __restrict__ mask,
    float* __restrict__ Out) {
  // 72 bf16 = 144 B row stride: 16B-aligned, 36-bank stagger (16 distinct rows)
  __shared__ bf16_t Ks[64][72];      // [key][d]   (async-filled)
  __shared__ bf16_t Vt[64][72];      // [d][key]   (transposed for PV B-frags)
  __shared__ bf16_t Ps[8][16][72];   // per-wave P tile [row][key]

  const int tid  = threadIdx.x;
  const int lane = tid & 31;
  const int wave = tid >> 5;
  const int qtile = blockIdx.x;
  const int h = blockIdx.y;
  const int b = blockIdx.z;
  const int qbase = qtile * 128;
  const int qrow0 = qbase + wave * 16;
  const float NEG_INF = -__builtin_inff();

  // Q fragments for this wave's 16x64 strip (2 K-steps of 32).
  v16bf qfrag[2];
  {
    const int m  = lane & 15;
    const int c0 = (lane < 16) ? 0 : 8;
    const bf16_t* p = Q + (size_t)(b * T_ + qrow0 + m) * H_ + h * HD_ + c0;
#pragma unroll
    for (int ks = 0; ks < 2; ++ks) {
      V16U u;
      u.h[0] = *(const v8bf*)(p + ks * 32);
      u.h[1] = *(const v8bf*)(p + ks * 32 + 16);
      qfrag[ks] = u.v;
    }
  }

  v8f o[4];
#pragma unroll
  for (int j = 0; j < 4; ++j) o[j] = (v8f){0, 0, 0, 0, 0, 0, 0, 0};
  float rmax[8], rsum[8];
#pragma unroll
  for (int r = 0; r < 8; ++r) { rmax[r] = NEG_INF; rsum[r] = 0.f; }

  const int kvr = tid >> 2, kvc = (tid & 3) * 16;   // cooperative 64x64 tile load
  const int nkt = qtile * 2 + 2;                    // causal key-tile bound

  for (int kt = 0; kt < nkt; ++kt) {
    const int kbase = kt * 64;
    {
      const unsigned grow =
          (unsigned)(((b * T_ + kbase + kvr) * H_ + h * HD_ + kvc) * 2);
      async_b128(lds_off(&Ks[kvr][kvc]),     grow,      K);   // K tile via DMA
      async_b128(lds_off(&Ks[kvr][kvc + 8]), grow + 16, K);
      const bf16_t* gv = V + (size_t)(b * T_ + kbase + kvr) * H_ + h * HD_ + kvc;
      v8bf v0 = *(const v8bf*)gv;
      v8bf v1 = *(const v8bf*)(gv + 8);
#pragma unroll
      for (int j = 0; j < 8; ++j) {                  // V transposed in registers
        Vt[kvc + j][kvr]     = v0[j];
        Vt[kvc + 8 + j][kvr] = v1[j];
      }
    }
    wait_async0();
    __syncthreads();

    // S = Q K^T : four 16x16 accumulators covering 64 keys.
    v8f s[4];
#pragma unroll
    for (int jt = 0; jt < 4; ++jt) {
      v8f a = (v8f){0, 0, 0, 0, 0, 0, 0, 0};
#pragma unroll
      for (int ks = 0; ks < 2; ++ks) {
        v16bf bf = load_b_frag(&Ks[jt * 16][ks * 32], 72, lane);
        a = wmma_bf16(qfrag[ks], bf, a);
      }
      s[jt] = a;
    }

    // scale + additive mask + causal mask
    const int m0 = qrow0 + ((lane < 16) ? 0 : 8);
#pragma unroll
    for (int jt = 0; jt < 4; ++jt) {
      const int n = kbase + jt * 16 + (lane & 15);
      const float mv = mask[b * T_ + n];
#pragma unroll
      for (int r = 0; r < 8; ++r) {
        float x = s[jt][r] * 0.125f + mv;    // HD^-0.5 = 1/8
        if (n > m0 + r) x = NEG_INF;
        s[jt][r] = x;
      }
    }

    // online softmax: per-row stats live in the 16-lane half owning that row
#pragma unroll
    for (int r = 0; r < 8; ++r) {
      float mx = fmaxf(fmaxf(s[0][r], s[1][r]), fmaxf(s[2][r], s[3][r]));
#pragma unroll
      for (int off = 8; off >= 1; off >>= 1)
        mx = fmaxf(mx, __shfl_xor(mx, off, 32));
      const float mnew = fmaxf(rmax[r], mx);
      const float corr = __expf(rmax[r] - mnew);
      rmax[r] = mnew;
      float ls = 0.f;
#pragma unroll
      for (int jt = 0; jt < 4; ++jt) {
        float p = __expf(s[jt][r] - mnew);
        s[jt][r] = p;
        ls += p;
      }
#pragma unroll
      for (int off = 8; off >= 1; off >>= 1) ls += __shfl_xor(ls, off, 32);
      rsum[r] = rsum[r] * corr + ls;
#pragma unroll
      for (int jt = 0; jt < 4; ++jt) o[jt][r] *= corr;
    }

    // C-layout -> A-layout via per-wave LDS round trip
    {
      const int nloc = lane & 15;
      const int roff = (lane < 16) ? 0 : 8;
#pragma unroll
      for (int jt = 0; jt < 4; ++jt)
#pragma unroll
        for (int r = 0; r < 8; ++r)
          Ps[wave][roff + r][jt * 16 + nloc] = (bf16_t)s[jt][r];
    }
    asm volatile("s_wait_dscnt 0" ::: "memory");  // wave-private: order store->load

    // O += P V
#pragma unroll
    for (int ks = 0; ks < 2; ++ks) {
      v16bf a = load_a_frag(&Ps[wave][0][ks * 32], 72, lane);
#pragma unroll
      for (int jt = 0; jt < 4; ++jt) {
        v16bf bf = load_b_frag(&Vt[jt * 16][ks * 32], 72, lane);
        o[jt] = wmma_bf16(a, bf, o[jt]);
      }
    }
    __syncthreads();
  }

  // epilogue: O /= rowsum, store fp32 [B,T,H]
  float inv[8];
#pragma unroll
  for (int r = 0; r < 8; ++r) inv[r] = 1.0f / rsum[r];
  const int m0 = qrow0 + ((lane < 16) ? 0 : 8);
#pragma unroll
  for (int jt = 0; jt < 4; ++jt) {
    const int d = jt * 16 + (lane & 15);
#pragma unroll
    for (int r = 0; r < 8; ++r)
      Out[(size_t)(b * T_ + m0 + r) * H_ + h * HD_ + d] = o[jt][r] * inv[r];
  }
}

// ---------------------------------------------------------------- launch
extern "C" void kernel_launch(void* const* d_in, const int* in_sizes, int n_in,
                              void* d_out, int out_size, void* d_ws, size_t ws_size,
                              hipStream_t stream) {
  const float* hidden = (const float*)d_in[0];
  const float* mask   = (const float*)d_in[1];
  const float* Wq = (const float*)d_in[2];
  const float* bq = (const float*)d_in[3];
  const float* Wk = (const float*)d_in[4];
  const float* bk = (const float*)d_in[5];
  const float* Wv = (const float*)d_in[6];
  const float* bv = (const float*)d_in[7];
  float* out = (float*)d_out;

  // workspace layout (bf16): X | Wq | Wk | Wv | Q | K | V   (~70 MB total)
  bf16_t* Xb  = (bf16_t*)d_ws;
  bf16_t* Wqb = Xb  + (size_t)BT_ * H_;
  bf16_t* Wkb = Wqb + (size_t)H_ * H_;
  bf16_t* Wvb = Wkb + (size_t)H_ * H_;
  bf16_t* Qb  = Wvb + (size_t)H_ * H_;
  bf16_t* Kb  = Qb  + (size_t)BT_ * H_;
  bf16_t* Vb  = Kb  + (size_t)BT_ * H_;

  {
    const int nx = BT_ * H_;
    cvt_f32_to_bf16<<<dim3((nx / 4 + 255) / 256), dim3(256), 0, stream>>>(hidden, Xb, nx);
    const int nw = H_ * H_;
    dim3 gw((nw / 4 + 255) / 256);
    cvt_f32_to_bf16<<<gw, dim3(256), 0, stream>>>(Wq, Wqb, nw);
    cvt_f32_to_bf16<<<gw, dim3(256), 0, stream>>>(Wk, Wkb, nw);
    cvt_f32_to_bf16<<<gw, dim3(256), 0, stream>>>(Wv, Wvb, nw);
  }

  qkv_gemm<<<dim3(BT_ / 128, H_ / 128, 3), dim3(256), 0, stream>>>(
      Xb, Wqb, Wkb, Wvb, bq, bk, bv, Qb, Kb, Vb);

  attn_fwd<<<dim3(T_ / 128, NH_, B_), dim3(256), 0, stream>>>(
      Qb, Kb, Vb, mask, out);
}
